// ProtoNetItemUserLL_54520314856137
// MI455X (gfx1250) — matrix operand order repaired and checked by reading, not compiled, not run on hardware
//
#include <hip/hip_runtime.h>
#include <hip/hip_bf16.h>

// ---------------------------------------------------------------------------
// Problem constants (from reference setup_inputs)
// ---------------------------------------------------------------------------
#define NQ 16384   // query rows
#define MM 2048    // memory slots
#define DD 128     // embedding dim
#define EPS 1e-8f

typedef __attribute__((ext_vector_type(16))) __bf16 v16bf;
typedef __attribute__((ext_vector_type(8)))  __bf16 v8bf;
typedef __attribute__((ext_vector_type(8)))  float  v8f;

// f32 -> bf16 round-to-nearest-even, returned as raw ushort bits
__device__ __forceinline__ unsigned short f2bf(float f) {
    unsigned u = __float_as_uint(f);
    unsigned r = (u + 0x7FFFu + ((u >> 16) & 1u)) >> 16;
    return (unsigned short)r;
}

__device__ __forceinline__ v16bf cat8(v8bf lo, v8bf hi) {
    return __builtin_shufflevector(lo, hi, 0,1,2,3,4,5,6,7,8,9,10,11,12,13,14,15);
}

// A-fragment (16x32 bf16, MxK): lane<16 -> M=lane, elems K0..15;
// lane>=16 -> M=lane-16, elems K16..31.  rowptr = base + M_row*ld.
__device__ __forceinline__ v16bf load_a_bf(const unsigned short* rowptr, int k0, int h) {
    v8bf lo = *(const v8bf*)(rowptr + k0 + h * 16);
    v8bf hi = *(const v8bf*)(rowptr + k0 + h * 16 + 8);
    return cat8(lo, hi);
}

// B-fragment (32x16 bf16, KxN) from row-major source whose rows are the N dim
// and contiguous dim is K: lane<16 -> N=lane, K=0..15; lane>=16 -> K=16..31.
__device__ __forceinline__ v16bf load_b_bf(const unsigned short* rowptr, int k0, int h) {
    return *(const v16bf*)(rowptr + k0 + h * 16);
}

__device__ __forceinline__ v8f wmma_bf16(v16bf a, v16bf b, v8f c) {
    return __builtin_amdgcn_wmma_f32_16x16x32_bf16(
        /*neg_a=*/false, a, /*neg_b=*/false, b,
        /*c_mod=*/(short)0, c, /*reuse_a=*/false, /*reuse_b=*/false);
}

// ---------------------------------------------------------------------------
// Prep kernels
// ---------------------------------------------------------------------------
__global__ void convert_bf16_kernel(const float* __restrict__ src,
                                    unsigned short* __restrict__ dst, int n) {
    int i = blockIdx.x * blockDim.x + threadIdx.x;
    if (i < n) dst[i] = f2bf(src[i]);
}

// One block (128 threads) per memory_index row: normalize, emit bf16.
__global__ __launch_bounds__(128)
void minorm_kernel(const float* __restrict__ mi, unsigned short* __restrict__ out) {
    int row = blockIdx.x;
    float v = mi[row * DD + threadIdx.x];
    float p = v * v;
    p += __shfl_xor(p, 1);  p += __shfl_xor(p, 2);  p += __shfl_xor(p, 4);
    p += __shfl_xor(p, 8);  p += __shfl_xor(p, 16);
    __shared__ float s[4];
    if ((threadIdx.x & 31) == 0) s[threadIdx.x >> 5] = p;
    __syncthreads();
    float tot = s[0] + s[1] + s[2] + s[3];
    float inv = 1.0f / fmaxf(sqrtf(tot), EPS);
    out[row * DD + threadIdx.x] = f2bf(v * inv);
}

// memory_w [M,D] f32  ->  mwT [D,M] bf16 (LDS-tiled transpose, 16x16 tiles)
__global__ __launch_bounds__(256)
void mwt_kernel(const float* __restrict__ mw, unsigned short* __restrict__ mwT) {
    __shared__ float tile[16][17];
    int tx = threadIdx.x & 15, ty = threadIdx.x >> 4;
    int m0 = blockIdx.x * 16, d0 = blockIdx.y * 16;
    tile[ty][tx] = mw[(m0 + ty) * DD + d0 + tx];
    __syncthreads();
    mwT[(d0 + ty) * MM + m0 + tx] = f2bf(tile[tx][ty]);
}

// ---------------------------------------------------------------------------
// Kernel 1: tq = q @ W_t^T + b_t ; tq_n = tq / max(||tq||, eps)  (bf16 out)
// 8 waves/block, one 16-row tile per wave.
// ---------------------------------------------------------------------------
__global__ __launch_bounds__(256)
void transform_norm_kernel(const unsigned short* __restrict__ qbf,   // [N,D] bf16
                           const unsigned short* __restrict__ wt,    // [D,D] bf16 (rows=out dim)
                           const float* __restrict__ b_t,            // [D]
                           unsigned short* __restrict__ tqn) {       // [N,D] bf16
    const int lane = threadIdx.x & 31;
    const int w    = threadIdx.x >> 5;
    const int c    = lane & 15;
    const int h    = lane >> 4;
    const int rbase = (blockIdx.x * 8 + w) * 16;

    v16bf A[4];
    const unsigned short* arow = qbf + (rbase + c) * DD;
    #pragma unroll
    for (int kk = 0; kk < 4; ++kk) A[kk] = load_a_bf(arow, kk * 32, h);

    v8f Ct[8];
    #pragma unroll
    for (int t = 0; t < 8; ++t) {
        const unsigned short* brow = wt + (16 * t + c) * DD;
        // batch the 4 B fragments, then issue the 4 WMMAs
        v16bf B[4];
        #pragma unroll
        for (int kk = 0; kk < 4; ++kk) B[kk] = load_b_bf(brow, kk * 32, h);
        v8f acc = {0.f,0.f,0.f,0.f,0.f,0.f,0.f,0.f};
        #pragma unroll
        for (int kk = 0; kk < 4; ++kk) acc = wmma_bf16(A[kk], B[kk], acc);
        float bias = b_t[16 * t + c];
        #pragma unroll
        for (int j = 0; j < 8; ++j) acc[j] += bias;
        Ct[t] = acc;
    }

    // row sums of squares: row = j + 8h, reduce across the 16 lanes of a half
    float p[8];
    #pragma unroll
    for (int j = 0; j < 8; ++j) {
        float s = 0.f;
        #pragma unroll
        for (int t = 0; t < 8; ++t) s += Ct[t][j] * Ct[t][j];
        s += __shfl_xor(s, 1); s += __shfl_xor(s, 2);
        s += __shfl_xor(s, 4); s += __shfl_xor(s, 8);
        p[j] = 1.0f / fmaxf(sqrtf(s), EPS);
    }

    #pragma unroll
    for (int t = 0; t < 8; ++t)
        #pragma unroll
        for (int j = 0; j < 8; ++j) {
            int row = rbase + j + 8 * h;
            tqn[row * DD + 16 * t + c] = f2bf(Ct[t][j] * p[j]);
        }
}

// ---------------------------------------------------------------------------
// Kernel 2: fused attention + softmax + weighted read + gating + blend.
// 4 waves/block, one 16-row query tile per wave, streaming over M in 32-chunks.
// cos_sim in [-1,1] => exp() is overflow-safe without max subtraction.
// ---------------------------------------------------------------------------
__global__ __launch_bounds__(128)
void fused_attn_kernel(const unsigned short* __restrict__ tqn,  // [N,D]  bf16
                       const unsigned short* __restrict__ mi,   // [M,D]  bf16 (normalized)
                       const unsigned short* __restrict__ mwT,  // [D,M]  bf16
                       const unsigned short* __restrict__ qbf,  // [N,D]  bf16
                       const unsigned short* __restrict__ wt2,  // [D,2D] bf16 (rows=out dim)
                       const float* __restrict__ query,         // [N,D]  f32
                       const float* __restrict__ b_t2,          // [D]
                       float* __restrict__ out,                 // [N,D]
                       float* __restrict__ memout) {            // [N,D]
    __shared__ unsigned short smem[4 * 2048];   // 4KB per wave, wave-private
    const int lane = threadIdx.x & 31;
    const int w    = threadIdx.x >> 5;
    const int c    = lane & 15;
    const int h    = lane >> 4;
    unsigned short* lds = smem + w * 2048;
    const int rbase = (blockIdx.x * 4 + w) * 16;

    // persistent A fragments of normalized queries (K = D = 128)
    v16bf A[4];
    const unsigned short* arow = tqn + (rbase + c) * DD;
    #pragma unroll
    for (int kk = 0; kk < 4; ++kk) A[kk] = load_a_bf(arow, kk * 32, h);

    v8f Cm[8];
    #pragma unroll
    for (int t = 0; t < 8; ++t) Cm[t] = (v8f){0.f,0.f,0.f,0.f,0.f,0.f,0.f,0.f};
    float sl[8] = {0.f,0.f,0.f,0.f,0.f,0.f,0.f,0.f};

    for (int cb = 0; cb < MM / 32; ++cb) {
        const int mbase = cb * 32;

        // prefetch next chunk's B data while this chunk computes (wave-uniform)
        if (cb + 1 < MM / 32) {
            const int nb = mbase + 32;
            __builtin_prefetch(mi  + (nb + lane) * DD, 0, 0);          // 32 rows of mi_n
            __builtin_prefetch(mwT + (lane * 4) * MM + nb, 0, 0);      // 32 of mwT's 128 rows (stride 4)
        }

        // --- S tiles (16 rows x 32 memory cols): batch 8 B frags, 8 WMMAs ---
        v16bf Bs[2][4];
        #pragma unroll
        for (int u = 0; u < 2; ++u) {
            const unsigned short* brow = mi + (mbase + 16 * u + c) * DD;
            #pragma unroll
            for (int kk = 0; kk < 4; ++kk) Bs[u][kk] = load_b_bf(brow, kk * 32, h);
        }
        #pragma unroll
        for (int u = 0; u < 2; ++u) {
            v8f S = {0.f,0.f,0.f,0.f,0.f,0.f,0.f,0.f};
            #pragma unroll
            for (int kk = 0; kk < 4; ++kk) S = wmma_bf16(A[kk], Bs[u][kk], S);
            #pragma unroll
            for (int j = 0; j < 8; ++j) {
                float e = __expf(S[j]);
                sl[j] += e;
                lds[(j + 8 * h) * 32 + 16 * u + c] = f2bf(e);   // P row-major 16x32
            }
        }
        // wave-private LDS: only need DS ops drained, no block barrier
        asm volatile("s_wait_dscnt 0x0" ::: "memory");

        // --- P as A-fragment; batch 8 mwT B frags, then 8 WMMAs ---
        v8bf plo = *(const v8bf*)(lds + c * 32 + h * 16);
        v8bf phi = *(const v8bf*)(lds + c * 32 + h * 16 + 8);
        v16bf Pa = cat8(plo, phi);
        v16bf Bv[8];
        #pragma unroll
        for (int t = 0; t < 8; ++t)
            Bv[t] = *(const v16bf*)(mwT + (16 * t + c) * MM + mbase + h * 16);
        #pragma unroll
        for (int t = 0; t < 8; ++t) Cm[t] = wmma_bf16(Pa, Bv[t], Cm[t]);

        asm volatile("s_wait_dscnt 0x0" ::: "memory");   // before next chunk overwrites P
    }

    // softmax denominators (reduce the 16 lanes of each half)
    #pragma unroll
    for (int j = 0; j < 8; ++j) {
        float s = sl[j];
        s += __shfl_xor(s, 1); s += __shfl_xor(s, 2);
        s += __shfl_xor(s, 4); s += __shfl_xor(s, 8);
        sl[j] = 1.0f / s;
    }

    // normalize, emit mem_emb, stage bf16 copy for gating GEMM
    #pragma unroll
    for (int t = 0; t < 8; ++t)
        #pragma unroll
        for (int j = 0; j < 8; ++j) {
            float m = Cm[t][j] * sl[j];
            Cm[t][j] = m;
            int row = rbase + j + 8 * h;
            memout[row * DD + 16 * t + c] = m;
            lds[(j + 8 * h) * DD + 16 * t + c] = f2bf(m);   // X mem-half 16x128
        }
    asm volatile("s_wait_dscnt 0x0" ::: "memory");

    // gating = tanh([mem_emb, query] @ W_t2^T + b_t2), K = 256
    v8f G[8];
    #pragma unroll
    for (int t = 0; t < 8; ++t) G[t] = (v8f){0.f,0.f,0.f,0.f,0.f,0.f,0.f,0.f};
    #pragma unroll
    for (int kk = 0; kk < 8; ++kk) {
        v16bf Ax;
        if (kk < 4) {   // mem_emb half of the concat, from LDS
            v8bf lo = *(const v8bf*)(lds + c * DD + kk * 32 + h * 16);
            v8bf hi = *(const v8bf*)(lds + c * DD + kk * 32 + h * 16 + 8);
            Ax = cat8(lo, hi);
        } else {        // query half of the concat, from global bf16
            Ax = load_a_bf(qbf + (rbase + c) * DD, (kk - 4) * 32, h);
        }
        // batch the 8 wt2 B fragments for this K-step, then 8 WMMAs
        v16bf Bw[8];
        #pragma unroll
        for (int t = 0; t < 8; ++t)
            Bw[t] = *(const v16bf*)(wt2 + (16 * t + c) * (2 * DD) + kk * 32 + h * 16);
        #pragma unroll
        for (int t = 0; t < 8; ++t) G[t] = wmma_bf16(Ax, Bw[t], G[t]);
    }

    // tanh gate + blend with original fp32 query
    #pragma unroll
    for (int t = 0; t < 8; ++t) {
        float bias = b_t2[16 * t + c];
        #pragma unroll
        for (int j = 0; j < 8; ++j) {
            float g = tanhf(G[t][j] + bias);
            int row = rbase + j + 8 * h;
            int col = 16 * t + c;
            float qv = query[row * DD + col];
            out[row * DD + col] = (1.0f - g) * qv + g * Cm[t][j];
        }
    }
}

// ---------------------------------------------------------------------------
// Launch
// ---------------------------------------------------------------------------
extern "C" void kernel_launch(void* const* d_in, const int* in_sizes, int n_in,
                              void* d_out, int out_size, void* d_ws, size_t ws_size,
                              hipStream_t stream) {
    const float* query = (const float*)d_in[0];   // [N,D]
    const float* W_t   = (const float*)d_in[1];   // [D,D]
    const float* b_t   = (const float*)d_in[2];   // [D]
    const float* miw   = (const float*)d_in[3];   // [M,D]
    const float* mw    = (const float*)d_in[4];   // [M,D]
    const float* W_t2  = (const float*)d_in[5];   // [D,2D]
    const float* b_t2  = (const float*)d_in[6];   // [D]

    float* out    = (float*)d_out;                // [N,D]
    float* memout = out + (size_t)NQ * DD;        // [N,D]

    char* ws = (char*)d_ws;
    unsigned short* q_bf   = (unsigned short*)ws;  ws += (size_t)NQ * DD * 2;   // 4 MB
    unsigned short* tqn    = (unsigned short*)ws;  ws += (size_t)NQ * DD * 2;   // 4 MB
    unsigned short* mi_bf  = (unsigned short*)ws;  ws += (size_t)MM * DD * 2;   // 512 KB
    unsigned short* mwT    = (unsigned short*)ws;  ws += (size_t)DD * MM * 2;   // 512 KB
    unsigned short* wt_bf  = (unsigned short*)ws;  ws += (size_t)DD * DD * 2;   // 32 KB
    unsigned short* wt2_bf = (unsigned short*)ws;  ws += (size_t)DD * 2 * DD * 2; // 64 KB

    // prep: converts / normalize / transpose
    convert_bf16_kernel<<<(NQ * DD + 255) / 256, 256, 0, stream>>>(query, q_bf, NQ * DD);
    convert_bf16_kernel<<<(DD * DD + 255) / 256, 256, 0, stream>>>(W_t, wt_bf, DD * DD);
    convert_bf16_kernel<<<(DD * 2 * DD + 255) / 256, 256, 0, stream>>>(W_t2, wt2_bf, DD * 2 * DD);
    minorm_kernel<<<MM, 128, 0, stream>>>(miw, mi_bf);
    mwt_kernel<<<dim3(MM / 16, DD / 16), 256, 0, stream>>>(mw, mwT);

    // transform + normalize queries
    transform_norm_kernel<<<NQ / 128, 256, 0, stream>>>(q_bf, wt_bf, b_t, tqn);

    // fused attention / read / gate / blend
    fused_attn_kernel<<<NQ / 64, 128, 0, stream>>>(tqn, mi_bf, mwT, q_bf, wt2_bf,
                                                   query, b_t2, out, memout);
}
